// SuperGlue_46677704573750
// MI455X (gfx1250) — matrix operand order repaired and checked by reading, not compiled, not run on hardware
//
#include <hip/hip_runtime.h>
#include <hip/hip_bf16.h>
#include <math.h>

typedef __attribute__((ext_vector_type(16))) _Float16 v16h;
typedef __attribute__((ext_vector_type(8)))  float    v8f;
typedef __attribute__((ext_vector_type(4)))  float    f32x4;

#define NPTS   2048
#define BATCH  2
#define ROWS   (BATCH*NPTS)   // 4096
#define DMODEL 256
#define NHEAD  4
#define DHEAD  64
#define NP1    2049
#define SINK_ITERS 100
#define LOG4096 8.317766166719343f
#define LOG2048 7.624618986159398f

// ---------------------------------------------------------------------------
// WMMA fragment helpers (CDNA5 16x16x32 f16 layouts, wave32)
// A 16x32: lane m=L&15, half=L>>4; VGPR v holds K=(v<4?0:16)+half*8+(v&3)*2
//   -> two contiguous float4 chunks at dword offsets half*4 and 8+half*4
// B^T rows [N][K]: lane n=L&15, half=L>>4; VGPR v holds K=(v<4?0:16)+half*8+(v&3)*2
//   -> two contiguous float4 chunks at dword offsets half*4 and 8+half*4
// ---------------------------------------------------------------------------
__device__ __forceinline__ v16h frag_make(f32x4 lo, f32x4 hi){
  union { f32x4 q[2]; v16h h; } u; u.q[0]=lo; u.q[1]=hi; return u.h;
}
__device__ __forceinline__ v16h load_fragA(const _Float16* row, int halfId){
  const f32x4* p = (const f32x4*)row;
  return frag_make(p[halfId], p[2+halfId]);
}
__device__ __forceinline__ v16h load_fragB(const _Float16* row, int halfId){
  const f32x4* p = (const f32x4*)row;
  return frag_make(p[halfId], p[2+halfId]);
}
__device__ __forceinline__ v8f zero8(){
  v8f z = {0.f,0.f,0.f,0.f,0.f,0.f,0.f,0.f}; return z;
}
#define WMMA(Af,Bf,Cf) __builtin_amdgcn_wmma_f32_16x16x32_f16(false,(Af),false,(Bf),(short)0,(Cf),false,false)

// ---------------------------------------------------------------------------
// Fallback WMMA GEMM (16x16 tile per wave) for N not multiple of 64.
// C[MxN] = scale * A[MxK] * B'[NxK]^T + bias
// ---------------------------------------------------------------------------
__global__ void gemm_wmma16(const _Float16* __restrict__ A, int lda,
                            const _Float16* __restrict__ B, int ldb,
                            const float* __restrict__ bias,
                            float* __restrict__ C, int ldc,
                            _Float16* __restrict__ C16, int ldc16, int c16T,
                            int M, int N, int K, float scale)
{
  const int tilesN = N >> 4;
  const int tiles  = (M >> 4) * tilesN;
  const int t = blockIdx.x * (blockDim.x >> 5) + (threadIdx.x >> 5);
  if (t >= tiles) return;                        // wave-uniform exit
  const int tm = t / tilesN, tn = t % tilesN;
  const int lane = threadIdx.x & 31;
  const int q = lane & 15, halfId = lane >> 4;
  const _Float16* Arow = A + (size_t)(tm*16 + q)*lda;
  const _Float16* Brow = B + (size_t)(tn*16 + q)*ldb;
  v8f acc = zero8();
  for (int k0 = 0; k0 < K; k0 += 32) {
    if (k0 + 32 < K) { __builtin_prefetch(Arow + k0 + 32, 0, 3);
                       __builtin_prefetch(Brow + k0 + 32, 0, 3); }
    v16h av = load_fragA(Arow + k0, halfId);
    v16h bv = load_fragB(Brow + k0, halfId);
    acc = WMMA(av, bv, acc);
  }
  const int col = tn*16 + q;
  const float bb = bias ? bias[col] : 0.f;
  #pragma unroll
  for (int r = 0; r < 8; ++r) {
    const int row = tm*16 + halfId*8 + r;
    const float val = acc[r]*scale + bb;
    if (C)   C[(size_t)row*ldc + col] = val;
    if (C16) {
      if (c16T) C16[(size_t)col*ldc16 + row] = (_Float16)val;
      else      C16[(size_t)row*ldc16 + col] = (_Float16)val;
    }
  }
}

// ---------------------------------------------------------------------------
// Main WMMA GEMM: 64x64 tile per wave -> 16 accumulators, 4x register reuse
// of both A and B fragments (16 WMMAs per 16 b128 loads per k-step).
// Requires M%64==0 && N%64==0 (true for all but kenc layer 1).
// ---------------------------------------------------------------------------
__global__ void gemm_wmma64(const _Float16* __restrict__ A, int lda,
                            const _Float16* __restrict__ B, int ldb,
                            const float* __restrict__ bias,
                            float* __restrict__ C, int ldc,
                            _Float16* __restrict__ C16, int ldc16, int c16T,
                            int M, int N, int K, float scale)
{
  const int tilesN = N >> 6;
  const int tiles  = (M >> 6) * tilesN;
  const int t = blockIdx.x * (blockDim.x >> 5) + (threadIdx.x >> 5);
  if (t >= tiles) return;                        // wave-uniform exit
  const int tm = (t / tilesN) * 64, tn = (t % tilesN) * 64;
  const int lane = threadIdx.x & 31;
  const int q = lane & 15, halfId = lane >> 4;
  const _Float16* Arow0 = A + (size_t)(tm + q)*lda;
  const _Float16* Arow1 = Arow0 + (size_t)16*lda;
  const _Float16* Arow2 = Arow0 + (size_t)32*lda;
  const _Float16* Arow3 = Arow0 + (size_t)48*lda;
  const _Float16* Brow0 = B + (size_t)(tn + q)*ldb;
  const _Float16* Brow1 = Brow0 + (size_t)16*ldb;
  const _Float16* Brow2 = Brow0 + (size_t)32*ldb;
  const _Float16* Brow3 = Brow0 + (size_t)48*ldb;
  v8f acc[16];
  #pragma unroll
  for (int i = 0; i < 16; ++i) acc[i] = zero8();
  for (int k0 = 0; k0 < K; k0 += 32) {
    v16h bv0 = load_fragB(Brow0 + k0, halfId);
    v16h bv1 = load_fragB(Brow1 + k0, halfId);
    v16h bv2 = load_fragB(Brow2 + k0, halfId);
    v16h bv3 = load_fragB(Brow3 + k0, halfId);
    v16h av;
    av = load_fragA(Arow0 + k0, halfId);
    acc[0]  = WMMA(av,bv0,acc[0]);  acc[1]  = WMMA(av,bv1,acc[1]);
    acc[2]  = WMMA(av,bv2,acc[2]);  acc[3]  = WMMA(av,bv3,acc[3]);
    av = load_fragA(Arow1 + k0, halfId);
    acc[4]  = WMMA(av,bv0,acc[4]);  acc[5]  = WMMA(av,bv1,acc[5]);
    acc[6]  = WMMA(av,bv2,acc[6]);  acc[7]  = WMMA(av,bv3,acc[7]);
    av = load_fragA(Arow2 + k0, halfId);
    acc[8]  = WMMA(av,bv0,acc[8]);  acc[9]  = WMMA(av,bv1,acc[9]);
    acc[10] = WMMA(av,bv2,acc[10]); acc[11] = WMMA(av,bv3,acc[11]);
    av = load_fragA(Arow3 + k0, halfId);
    acc[12] = WMMA(av,bv0,acc[12]); acc[13] = WMMA(av,bv1,acc[13]);
    acc[14] = WMMA(av,bv2,acc[14]); acc[15] = WMMA(av,bv3,acc[15]);
  }
  float bb[4];
  #pragma unroll
  for (int j = 0; j < 4; ++j) bb[j] = bias ? bias[tn + j*16 + q] : 0.f;
  #pragma unroll
  for (int i = 0; i < 4; ++i) {
    #pragma unroll
    for (int j = 0; j < 4; ++j) {
      const int col = tn + j*16 + q;
      #pragma unroll
      for (int r = 0; r < 8; ++r) {
        const int row = tm + i*16 + halfId*8 + r;
        const float val = acc[i*4+j][r]*scale + bb[j];
        if (C)   C[(size_t)row*ldc + col] = val;
        if (C16) {
          if (c16T) C16[(size_t)col*ldc16 + row] = (_Float16)val;
          else      C16[(size_t)row*ldc16 + col] = (_Float16)val;
        }
      }
    }
  }
}

// ---------------------------------------------------------------------------
// Fused flash attention: one wave = 32 query rows (2 row-tiles) of one (b,h);
// K/V fragments are loaded once per 32-key step and reused by both row-tiles.
// Qh,Kh: f16 [B*N][256] (head slice at h*64). Vt: f16 [256][B*N] (transposed).
// Online softmax; P routed via LDS (C-layout -> A-layout).
// ---------------------------------------------------------------------------
__global__ void flash_attn_k(const _Float16* __restrict__ Qh,
                             const _Float16* __restrict__ Kh,
                             const _Float16* __restrict__ Vt,
                             _Float16* __restrict__ Msg)
{
  __shared__ _Float16 Pl[2*16*32];
  const int rb = blockIdx.x, h = blockIdx.y, b = blockIdx.z;
  const int lane = threadIdx.x & 31;
  const int q = lane & 15, halfId = lane >> 4;
  v16h qa[2][2];
  #pragma unroll
  for (int i = 0; i < 2; ++i) {
    const _Float16* Qrow = Qh + (size_t)(b*NPTS + rb*32 + i*16 + q)*DMODEL + h*DHEAD;
    qa[i][0] = load_fragA(Qrow,      halfId);
    qa[i][1] = load_fragA(Qrow + 32, halfId);
  }
  v8f o[2][4];
  float mr[2][8], lr[2][8];
  #pragma unroll
  for (int i = 0; i < 2; ++i){
    #pragma unroll
    for (int jb = 0; jb < 4; ++jb) o[i][jb] = zero8();
    #pragma unroll
    for (int r = 0; r < 8; ++r) { mr[i][r] = -1e30f; lr[i][r] = 0.f; }
  }

  for (int j0 = 0; j0 < NPTS; j0 += 32) {
    const _Float16* Kr0 = Kh + (size_t)(b*NPTS + j0      + q)*DMODEL + h*DHEAD;
    const _Float16* Kr1 = Kh + (size_t)(b*NPTS + j0 + 16 + q)*DMODEL + h*DHEAD;
    const v16h kb0 = load_fragB(Kr0,      halfId);
    const v16h kb1 = load_fragB(Kr0 + 32, halfId);
    const v16h kb2 = load_fragB(Kr1,      halfId);
    const v16h kb3 = load_fragB(Kr1 + 32, halfId);
    #pragma unroll
    for (int i = 0; i < 2; ++i) {
      v8f s0 = zero8(), s1 = zero8();
      s0 = WMMA(qa[i][0], kb0, s0); s0 = WMMA(qa[i][1], kb1, s0);
      s1 = WMMA(qa[i][0], kb2, s1); s1 = WMMA(qa[i][1], kb3, s1);
      #pragma unroll
      for (int r = 0; r < 8; ++r) {
        float a = s0[r]*0.0625f, c = s1[r]*0.0625f;   // /sqrt(D), D=256
        float mx = fmaxf(a, c);
        mx = fmaxf(mx, __shfl_xor(mx, 1, 32));
        mx = fmaxf(mx, __shfl_xor(mx, 2, 32));
        mx = fmaxf(mx, __shfl_xor(mx, 4, 32));
        mx = fmaxf(mx, __shfl_xor(mx, 8, 32));        // row-reduce within 16-lane half
        const float nm = fmaxf(mr[i][r], mx);
        const float f  = __expf(mr[i][r] - nm);
        mr[i][r] = nm; lr[i][r] *= f;
        o[i][0][r] *= f; o[i][1][r] *= f; o[i][2][r] *= f; o[i][3][r] *= f;
        const float p0 = __expf(a - nm), p1 = __expf(c - nm);
        float rs = p0 + p1;
        rs += __shfl_xor(rs, 1, 32);
        rs += __shfl_xor(rs, 2, 32);
        rs += __shfl_xor(rs, 4, 32);
        rs += __shfl_xor(rs, 8, 32);
        lr[i][r] += rs;
        const int rowl = halfId*8 + r;
        Pl[i*512 + rowl*32 + q]      = (_Float16)p0;
        Pl[i*512 + rowl*32 + 16 + q] = (_Float16)p1;
      }
    }
    __syncthreads();
    const v16h pa0 = load_fragA(Pl +       q*32, halfId);
    const v16h pa1 = load_fragA(Pl + 512 + q*32, halfId);
    #pragma unroll
    for (int jb = 0; jb < 4; ++jb) {
      const _Float16* Vb = Vt + (size_t)(h*DHEAD + jb*16 + q)*ROWS + b*NPTS + j0;
      const v16h vb = load_fragB(Vb, halfId);
      o[0][jb] = WMMA(pa0, vb, o[0][jb]);
      o[1][jb] = WMMA(pa1, vb, o[1][jb]);
    }
    __syncthreads();
  }
  #pragma unroll
  for (int i = 0; i < 2; ++i) {
    #pragma unroll
    for (int r = 0; r < 8; ++r) {
      const float inv = 1.f / lr[i][r];
      const size_t row = (size_t)(b*NPTS + rb*32 + i*16 + halfId*8 + r);
      _Float16* op = Msg + row*DMODEL + h*DHEAD + q;
      op[0]  = (_Float16)(o[i][0][r]*inv);
      op[16] = (_Float16)(o[i][1][r]*inv);
      op[32] = (_Float16)(o[i][2][r]*inv);
      op[48] = (_Float16)(o[i][3][r]*inv);
    }
  }
}

// ------------------------------ small kernels ------------------------------
__global__ void cvt16_k(const float* __restrict__ S, int sld, int cols,
                        _Float16* __restrict__ D, int dld, int rows){
  int t = blockIdx.x*blockDim.x + threadIdx.x;
  if (t >= rows*dld) return;
  int r = t / dld, c = t % dld;
  D[t] = (c < cols) ? (_Float16)S[(size_t)r*sld + c] : (_Float16)0.f;
}
__global__ void copy16_k(const _Float16* __restrict__ S, int sld,
                         _Float16* __restrict__ D, int dld, int rows, int cols){
  int t = blockIdx.x*blockDim.x + threadIdx.x;
  if (t >= rows*cols) return;
  int r = t / cols, c = t % cols;
  D[(size_t)r*dld + c] = S[(size_t)r*sld + c];
}
__global__ void bn_stats_k(const float* __restrict__ X, int ld, int rows,
                           float* __restrict__ stats){
  const int c = blockIdx.x, tid = threadIdx.x;
  float s = 0.f, sq = 0.f;
  for (int r = tid; r < rows; r += blockDim.x){
    float v = X[(size_t)r*ld + c]; s += v; sq += v*v;
  }
  __shared__ float s1[256], s2[256];
  s1[tid] = s; s2[tid] = sq; __syncthreads();
  for (int off = 128; off; off >>= 1){
    if (tid < off){ s1[tid] += s1[tid+off]; s2[tid] += s2[tid+off]; }
    __syncthreads();
  }
  if (!tid){
    float mean = s1[0]/rows;
    stats[2*c] = mean; stats[2*c+1] = s2[0]/rows - mean*mean;
  }
}
__global__ void bn_apply_k(const float* __restrict__ X, int ld,
                           const float* __restrict__ stats,
                           const float* __restrict__ g, const float* __restrict__ beta,
                           _Float16* __restrict__ Y, int yld, int rows, int cols){
  int t = blockIdx.x*blockDim.x + threadIdx.x;
  if (t >= rows*cols) return;
  int r = t / cols, c = t % cols;
  float v = (X[(size_t)r*ld + c] - stats[2*c]) * rsqrtf(stats[2*c+1] + 1e-5f) * g[c] + beta[c];
  Y[(size_t)r*yld + c] = (_Float16)fmaxf(v, 0.f);
}
__global__ void kenc_in_k(const float* __restrict__ kp, const float* __restrict__ sc,
                          const int* __restrict__ ih, const int* __restrict__ iw,
                          _Float16* __restrict__ D){
  int t = blockIdx.x*blockDim.x + threadIdx.x;
  if (t >= ROWS) return;
  const float w = (float)iw[0], h = (float)ih[0];
  const float s = fmaxf(w, h) * 0.7f;
  _Float16* d = D + (size_t)t*32;
  d[0] = (_Float16)((kp[2*t]   - w*0.5f)/s);
  d[1] = (_Float16)((kp[2*t+1] - h*0.5f)/s);
  d[2] = (_Float16)sc[t];
  for (int c = 3; c < 32; ++c) d[c] = (_Float16)0.f;
}
__global__ void add_desc_k(const float* __restrict__ desc, const float* __restrict__ enc,
                           float* __restrict__ act, _Float16* __restrict__ actH){
  size_t t = (size_t)blockIdx.x*blockDim.x + threadIdx.x;
  if (t >= (size_t)ROWS*DMODEL) return;
  int b = t/(NPTS*DMODEL); int rem = t%(NPTS*DMODEL);
  int i = rem/DMODEL, d = rem%DMODEL;
  float v = desc[(size_t)b*DMODEL*NPTS + (size_t)d*NPTS + i] + enc[t];
  act[t] = v; actH[t] = (_Float16)v;
}
__global__ void bins_k(float* __restrict__ Z, const float* __restrict__ alpha){
  int t = blockIdx.x*blockDim.x + threadIdx.x;
  if (t >= BATCH*(NP1+NPTS)) return;
  const float a = alpha[0];
  int b = t/(NP1+NPTS), r = t%(NP1+NPTS);
  float* Zb = Z + (size_t)b*NP1*NP1;
  if (r < NP1) Zb[(size_t)NPTS*NP1 + r] = a;           // bottom row (incl. corner)
  else         Zb[(size_t)(r-NP1)*NP1 + NPTS] = a;     // right column
}
__global__ void transpose_k(const float* __restrict__ S, float* __restrict__ D){
  __shared__ float tile[32][33];
  const int b = blockIdx.z;
  const float* Sb = S + (size_t)b*NP1*NP1;
  float* Db = D + (size_t)b*NP1*NP1;
  int x = blockIdx.x*32 + threadIdx.x;
  for (int i = 0; i < 32; i += 8){
    int y = blockIdx.y*32 + threadIdx.y + i;
    if (x < NP1 && y < NP1) tile[threadIdx.y+i][threadIdx.x] = Sb[(size_t)y*NP1 + x];
  }
  __syncthreads();
  int x2 = blockIdx.y*32 + threadIdx.x;
  for (int i = 0; i < 32; i += 8){
    int y2 = blockIdx.x*32 + threadIdx.y + i;
    if (x2 < NP1 && y2 < NP1) Db[(size_t)y2*NP1 + x2] = tile[threadIdx.x][threadIdx.y+i];
  }
}
__global__ void zero_k(float* __restrict__ p, int n){
  int t = blockIdx.x*blockDim.x + threadIdx.x;
  if (t < n) p[t] = 0.f;
}
// out[b][i] = target(i) - logsumexp_j( Z[b][i][j] + in[b][j] )   (row-coalesced)
__global__ void sink_k(const float* __restrict__ Z, const float* __restrict__ in,
                       float* __restrict__ out){
  const int i = blockIdx.x, b = blockIdx.y, tid = threadIdx.x;
  const float* row = Z + ((size_t)b*NP1 + i)*NP1;
  const float* vv = in + (size_t)b*NP1;
  float m = -1e30f, s = 0.f;
  for (int j = tid; j < NP1; j += 256){
    float z = row[j] + vv[j];
    if (z > m){ s = s*__expf(m - z) + 1.f; m = z; }
    else       s += __expf(z - m);
  }
  __shared__ float sm[256], ss[256];
  sm[tid] = m; ss[tid] = s; __syncthreads();
  for (int off = 128; off; off >>= 1){
    if (tid < off){
      float m2 = sm[tid+off], s2 = ss[tid+off];
      float M = fmaxf(sm[tid], m2);
      ss[tid] = ss[tid]*__expf(sm[tid]-M) + s2*__expf(m2-M);
      sm[tid] = M;
    }
    __syncthreads();
  }
  if (!tid){
    const float target = (i < NPTS) ? -LOG4096 : (LOG2048 - LOG4096);
    out[(size_t)b*NP1 + i] = target - (sm[0] + __logf(ss[0]));
  }
}
// batch-0 argmax over j<2048 of Z[i][j]+addv[j]; score += ownv[i] + log(4096)
__global__ void rowmax_k(const float* __restrict__ Z, const float* __restrict__ addv,
                         const float* __restrict__ ownv,
                         float* __restrict__ msc, int* __restrict__ idx){
  const int i = blockIdx.x, tid = threadIdx.x;
  const float* row = Z + (size_t)i*NP1;
  float best = -1e30f; int bi = 0;
  for (int j = tid; j < NPTS; j += 256){
    float z = row[j] + addv[j];
    if (z > best){ best = z; bi = j; }
  }
  __shared__ float sv[256]; __shared__ int si[256];
  sv[tid] = best; si[tid] = bi; __syncthreads();
  for (int off = 128; off; off >>= 1){
    if (tid < off && sv[tid+off] > sv[tid]){ sv[tid] = sv[tid+off]; si[tid] = si[tid+off]; }
    __syncthreads();
  }
  if (!tid){ msc[i] = sv[0] + ownv[i] + LOG4096; idx[i] = si[0]; }
}
__global__ void match_k(const int* __restrict__ id0, const int* __restrict__ id1,
                        const float* __restrict__ ms0, const float* __restrict__ ms1,
                        float* __restrict__ out){
  int i = blockIdx.x*blockDim.x + threadIdx.x;
  if (i >= NPTS) return;
  int j = id0[i];
  out[i]        = (id1[j] == i && ms0[i] > 0.2f) ? (float)j  : -1.f;
  out[4096 + i] = ms0[i];
  int j2 = id1[i];
  out[2048 + i] = (id0[j2] == i && ms1[i] > 0.2f) ? (float)j2 : -1.f;
  out[6144 + i] = ms1[i];
}

// ---------------------------------------------------------------------------
extern "C" void kernel_launch(void* const* d_in, const int* in_sizes, int n_in,
                              void* d_out, int out_size, void* d_ws, size_t ws_size,
                              hipStream_t stream)
{
  (void)in_sizes; (void)n_in; (void)out_size; (void)ws_size;
  // ---- inputs (setup_inputs dict order, params flattened recursively) ----
  const float* kpts0 = (const float*)d_in[0];
  const float* kpts1 = (const float*)d_in[1];
  const float* desc0 = (const float*)d_in[2];
  const float* desc1 = (const float*)d_in[3];
  const float* sc0   = (const float*)d_in[4];
  const float* sc1   = (const float*)d_in[5];
  const int*   ih    = (const int*)d_in[6];
  const int*   iw    = (const int*)d_in[7];
  int pi = 8;
  const float *kw[4], *kb[4], *kg[3], *kbt[3];
  for (int i = 0; i < 4; ++i){
    kw[i] = (const float*)d_in[pi++]; kb[i] = (const float*)d_in[pi++];
    if (i < 3){ kg[i] = (const float*)d_in[pi++]; kbt[i] = (const float*)d_in[pi++]; }
  }
  const float* gp[6][14];
  for (int L = 0; L < 6; ++L)
    for (int j = 0; j < 14; ++j) gp[L][j] = (const float*)d_in[pi++];
  const float* finw = (const float*)d_in[pi++];
  const float* finb = (const float*)d_in[pi++];
  const float* bins = (const float*)d_in[pi++];

  // ---- workspace carve (~105 MB) ----
  char* wp = (char*)d_ws;
  auto take = [&](size_t bytes)->void*{ void* p = wp; wp += (bytes + 255) & ~(size_t)255; return p; };
  float*    act0  = (float*)take((size_t)ROWS*DMODEL*4);
  float*    act1  = (float*)take((size_t)ROWS*DMODEL*4);
  _Float16* actH0 = (_Float16*)take((size_t)ROWS*DMODEL*2);
  _Float16* actH1 = (_Float16*)take((size_t)ROWS*DMODEL*2);
  _Float16* qh    = (_Float16*)take((size_t)ROWS*DMODEL*2);
  _Float16* kh    = (_Float16*)take((size_t)ROWS*DMODEL*2);
  _Float16* vt    = (_Float16*)take((size_t)ROWS*DMODEL*2);   // [256][4096]
  _Float16* msg   = (_Float16*)take((size_t)ROWS*DMODEL*2);
  _Float16* h16a  = (_Float16*)take((size_t)ROWS*512*2);
  _Float16* wbuf  = (_Float16*)take((size_t)512*512*2);
  _Float16* kin   = (_Float16*)take((size_t)ROWS*32*2);
  _Float16* m0h   = (_Float16*)take((size_t)ROWS*DMODEL*2);
  _Float16* m1h   = (_Float16*)take((size_t)ROWS*DMODEL*2);
  float*    t0    = (float*)take((size_t)ROWS*512*4);
  float*    t1    = (float*)take((size_t)ROWS*DMODEL*4);
  float*    stats = (float*)take((size_t)512*2*4);
  float*    Zc    = (float*)take((size_t)BATCH*NP1*NP1*4);
  float*    ZcT   = (float*)take((size_t)BATCH*NP1*NP1*4);
  float*    uvec  = (float*)take((size_t)BATCH*NP1*4);
  float*    vvec  = (float*)take((size_t)BATCH*NP1*4);
  float*    ms0   = (float*)take((size_t)NPTS*4);
  float*    ms1   = (float*)take((size_t)NPTS*4);
  int*      id0   = (int*)take((size_t)NPTS*4);
  int*      id1   = (int*)take((size_t)NPTS*4);

  auto cvt = [&](const float* S, int sld, int cols, _Float16* D, int dld, int rows){
    int n = rows*dld;
    cvt16_k<<<(n+255)/256, 256, 0, stream>>>(S, sld, cols, D, dld, rows);
  };
  auto gemm = [&](const _Float16* A, int lda, const _Float16* B, int ldb,
                  const float* bias, float* C, int ldc,
                  _Float16* C16, int ldc16, int tr, int M, int N, int K, float sc){
    if ((M & 63) == 0 && (N & 63) == 0) {
      int tiles = (M/64)*(N/64);
      gemm_wmma64<<<(tiles+3)/4, 128, 0, stream>>>(A,lda,B,ldb,bias,C,ldc,C16,ldc16,tr,M,N,K,sc);
    } else {
      int tiles = (M/16)*(N/16);
      gemm_wmma16<<<(tiles+3)/4, 128, 0, stream>>>(A,lda,B,ldb,bias,C,ldc,C16,ldc16,tr,M,N,K,sc);
    }
  };

  // ---- keypoint encoder (run per image; BN stats are per call) ----
  auto kenc = [&](const float* kp, const float* sc, const float* desc,
                  float* act, _Float16* actH){
    kenc_in_k<<<(ROWS+255)/256, 256, 0, stream>>>(kp, sc, ih, iw, kin);
    cvt(kw[0], 3, 3, wbuf, 32, 32);                                    // pad K 3->32
    gemm(kin, 32, wbuf, 32, kb[0], t0, 32, nullptr, 0, 0, ROWS, 32, 32, 1.f);
    bn_stats_k<<<32, 256, 0, stream>>>(t0, 32, ROWS, stats);
    bn_apply_k<<<(ROWS*32+255)/256, 256, 0, stream>>>(t0, 32, stats, kg[0], kbt[0], h16a, 32, ROWS, 32);
    cvt(kw[1], 32, 32, wbuf, 32, 64);
    gemm(h16a, 32, wbuf, 32, kb[1], t0, 64, nullptr, 0, 0, ROWS, 64, 32, 1.f);
    bn_stats_k<<<64, 256, 0, stream>>>(t0, 64, ROWS, stats);
    bn_apply_k<<<(ROWS*64+255)/256, 256, 0, stream>>>(t0, 64, stats, kg[1], kbt[1], h16a, 64, ROWS, 64);
    cvt(kw[2], 64, 64, wbuf, 64, 128);
    gemm(h16a, 64, wbuf, 64, kb[2], t0, 128, nullptr, 0, 0, ROWS, 128, 64, 1.f);
    bn_stats_k<<<128, 256, 0, stream>>>(t0, 128, ROWS, stats);
    bn_apply_k<<<(ROWS*128+255)/256, 256, 0, stream>>>(t0, 128, stats, kg[2], kbt[2], h16a, 128, ROWS, 128);
    cvt(kw[3], 128, 128, wbuf, 128, 256);
    gemm(h16a, 128, wbuf, 128, kb[3], t1, 256, nullptr, 0, 0, ROWS, 256, 128, 1.f);
    add_desc_k<<<((unsigned)((size_t)ROWS*DMODEL+255)/256), 256, 0, stream>>>(desc, t1, act, actH);
  };
  kenc(kpts0, sc0, desc0, act0, actH0);
  kenc(kpts1, sc1, desc1, act1, actH1);

  // ---- GNN attention layers ----
  auto attn = [&](const float* const* P, int dst, int src){
    float*    Aq  = dst ? act1  : act0;
    _Float16* AqH = dst ? actH1 : actH0;
    _Float16* AsH = src ? actH1 : actH0;
    cvt(P[0], 256, 256, wbuf, 256, 256);
    gemm(AqH, 256, wbuf, 256, P[1], nullptr, 0, qh, 256, 0, ROWS, 256, 256, 1.f);
    cvt(P[2], 256, 256, wbuf, 256, 256);
    gemm(AsH, 256, wbuf, 256, P[3], nullptr, 0, kh, 256, 0, ROWS, 256, 256, 1.f);
    cvt(P[4], 256, 256, wbuf, 256, 256);
    gemm(AsH, 256, wbuf, 256, P[5], nullptr, 0, vt, ROWS, 1, ROWS, 256, 256, 1.f); // V^T
    flash_attn_k<<<dim3(NPTS/32, NHEAD, BATCH), 32, 0, stream>>>(qh, kh, vt, msg);
    cvt(P[6], 256, 256, wbuf, 256, 256);
    gemm(msg, 256, wbuf, 256, P[7], nullptr, 0, h16a + 256, 512, 0, ROWS, 256, 256, 1.f);
    copy16_k<<<(ROWS*256+255)/256, 256, 0, stream>>>(AqH, 256, h16a, 512, ROWS, 256); // concat
    cvt(P[8], 512, 512, wbuf, 512, 512);
    gemm(h16a, 512, wbuf, 512, P[9], t0, 512, nullptr, 0, 0, ROWS, 512, 512, 1.f);
    bn_stats_k<<<512, 256, 0, stream>>>(t0, 512, ROWS, stats);
    bn_apply_k<<<((unsigned)((size_t)ROWS*512+255)/256), 256, 0, stream>>>(t0, 512, stats, P[10], P[11], h16a, 512, ROWS, 512);
    cvt(P[12], 512, 512, wbuf, 512, 256);
    gemm(h16a, 512, wbuf, 512, P[13], Aq, 256, AqH, 256, 0, ROWS, 256, 512, 1.f);
  };
  for (int L = 0; L < 6; ++L){
    if ((L & 1) == 0){ attn(gp[L], 0, 0); attn(gp[L], 1, 1); }   // self
    else             { attn(gp[L], 0, 1); attn(gp[L], 1, 0); }   // cross (d1 uses updated d0)
  }

  // ---- final projection + score matrix (scale 1/sqrt(256)) ----
  cvt(finw, 256, 256, wbuf, 256, 256);
  gemm(actH0, 256, wbuf, 256, finb, nullptr, 0, m0h, 256, 0, ROWS, 256, 256, 1.f);
  gemm(actH1, 256, wbuf, 256, finb, nullptr, 0, m1h, 256, 0, ROWS, 256, 256, 1.f);
  for (int b = 0; b < BATCH; ++b)
    gemm(m0h + (size_t)b*NPTS*256, 256, m1h + (size_t)b*NPTS*256, 256, nullptr,
         Zc + (size_t)b*NP1*NP1, NP1, nullptr, 0, 0, NPTS, NPTS, 256, 0.0625f);
  bins_k<<<(BATCH*(NP1+NPTS)+255)/256, 256, 0, stream>>>(Zc, bins);
  transpose_k<<<dim3(65, 65, BATCH), dim3(32, 8), 0, stream>>>(Zc, ZcT);

  // ---- Sinkhorn (Z constant; Z and Z^T give coalesced row LSE both ways) ----
  zero_k<<<(BATCH*NP1+255)/256, 256, 0, stream>>>(uvec, BATCH*NP1);
  zero_k<<<(BATCH*NP1+255)/256, 256, 0, stream>>>(vvec, BATCH*NP1);
  for (int it = 0; it < SINK_ITERS; ++it){
    sink_k<<<dim3(NP1, BATCH), 256, 0, stream>>>(Zc,  vvec, uvec);
    sink_k<<<dim3(NP1, BATCH), 256, 0, stream>>>(ZcT, uvec, vvec);
  }

  // ---- matching (batch 0) ----
  rowmax_k<<<NPTS, 256, 0, stream>>>(Zc,  vvec, uvec, ms0, id0);
  rowmax_k<<<NPTS, 256, 0, stream>>>(ZcT, uvec, vvec, ms1, id1);
  match_k<<<(NPTS+255)/256, 256, 0, stream>>>(id0, id1, ms0, ms1, (float*)d_out);
}
